// RealNVPFlow_1434519077072
// MI455X (gfx1250) — compile-verified, matrix operand order checked by reading, not run on hardware
//
#include <hip/hip_runtime.h>

#define BATCH   8192
#define NDIM    4096
#define HDIM    256
#define NLAYERS 8
#define NMASK   2048
#define TILE_B  16

typedef __attribute__((ext_vector_type(16))) __bf16 v16bf;
typedef __attribute__((ext_vector_type(8)))  __bf16 v8bf;
typedef __attribute__((ext_vector_type(8)))  float  v8f;

// ---------------------------------------------------------------------------
// Pack fp32 weights (K x Nc, row-major, per layer) into bf16 WMMA B-fragments.
// Fragment (kt, nt) = 32x16 block, stored as 512 bf16: lane l holds 16
// contiguous bf16 = cols n = nt*16 + (l&15), rows K in {kb..kb+7, kb+16..kb+23}
// with kb = (l>>4)*8  (matches V_WMMA_*_16X16X32 8-VGPR 16-bit B layout).
// ---------------------------------------------------------------------------
__global__ void pack_weights(const float* __restrict__ W,
                             unsigned short* __restrict__ dst,
                             int layers, int K, int Nc) {
  long long tid   = (long long)blockIdx.x * blockDim.x + threadIdx.x;
  long long total = (long long)layers * K * Nc;
  if (tid >= total) return;
  int KN     = K * Nc;
  int layer  = (int)(tid / KN);
  int rem    = (int)(tid - (long long)layer * KN);
  int frag   = rem >> 9;        // 512 elems per fragment
  int within = rem & 511;
  int lane   = within >> 4;
  int e      = within & 15;
  int NT     = Nc >> 4;
  int kt     = frag / NT;
  int nt     = frag - kt * NT;
  int n      = (nt << 4) + (lane & 15);
  int kb     = (lane >> 4) << 3;
  int k      = (kt << 5) + kb + (e < 8 ? e : e + 8);
  unsigned int u = __float_as_uint(W[(long long)layer * KN + (long long)k * Nc + n]);
  // round-to-nearest-even fp32 -> bf16
  dst[tid] = (unsigned short)((u + 0x7FFFu + ((u >> 16) & 1u)) >> 16);
}

__device__ __forceinline__ unsigned int pack_bf16x2(float a, float b) {
  union { __bf16 h[2]; unsigned int u; } v;
  v.h[0] = (__bf16)a;
  v.h[1] = (__bf16)b;
  return v.u;   // lowers to v_cvt_pk_bf16_f32
}

// Build a 16x32 bf16 A-fragment from a row-major bf16 LDS matrix [16][kdim].
__device__ __forceinline__ v16bf load_a_lds(const __bf16* h, int kdim,
                                            int arow, int kb, int kt) {
  const v8bf* plo = (const v8bf*)(h + arow * kdim + (kt << 5) + kb);
  const v8bf* phi = (const v8bf*)(h + arow * kdim + (kt << 5) + kb + 16);
  v8bf lo = *plo, hi = *phi;
  v16bf a;
#pragma unroll
  for (int e = 0; e < 8; ++e) { a[e] = lo[e]; a[e + 8] = hi[e]; }
  return a;
}

#define WMMA_BF16(A, Bf, C) \
  __builtin_amdgcn_wmma_f32_16x16x32_bf16(false, (A), false, (Bf), (short)0, (C), false, false)

// ---------------------------------------------------------------------------
// Whole 8-layer RealNVP flow for a 16-row batch tile, x resident in LDS.
// ---------------------------------------------------------------------------
__launch_bounds__(256)
__global__ void realnvp_flow(const float* __restrict__ z,
                             const float* __restrict__ b1,
                             const float* __restrict__ b2,
                             const float* __restrict__ b3,
                             const unsigned short* __restrict__ W1u,
                             const unsigned short* __restrict__ W2u,
                             const unsigned short* __restrict__ W3u,
                             float* __restrict__ x_out,
                             float* __restrict__ ld_out) {
  extern __shared__ char smem_raw[];
  float*  xs    = (float*)smem_raw;                                 // 16 x 4096 fp32
  __bf16* xmb   = (__bf16*)(smem_raw + (size_t)TILE_B * NDIM * 4);  // 32 frags * 512 bf16
  __bf16* h1    = xmb + 32 * 512;                                   // 16 x 256 bf16
  __bf16* h2    = h1 + TILE_B * HDIM;                               // 16 x 256 bf16
  float*  ldacc = (float*)(h2 + TILE_B * HDIM);                     // 16 fp32

  const int tid   = threadIdx.x;
  const int wave  = tid >> 5;
  const int lane  = tid & 31;
  const int col   = lane & 15;          // N-column within tile / A-row
  const int rowhi = (lane >> 4) << 3;   // D-tile row offset (0 or 8)
  const int kb    = (lane >> 4) << 3;   // A/B K sub-offset (0 or 8)
  const int b0    = blockIdx.x * TILE_B;

  const __bf16* W1p = (const __bf16*)W1u;
  const __bf16* W2p = (const __bf16*)W2u;
  const __bf16* W3p = (const __bf16*)W3u;

  // ---- stage x tile: 16 rows of z -> LDS (fp32) ----
  {
    const float4* src = (const float4*)(z + (size_t)b0 * NDIM);
    float4*       dst = (float4*)xs;
    for (int it = tid; it < TILE_B * NDIM / 4; it += 256) dst[it] = src[it];
  }
  if (tid < TILE_B) ldacc[tid] = 0.0f;
  __syncthreads();

  float ldp[8];
#pragma unroll
  for (int g = 0; g < 8; ++g) ldp[g] = 0.0f;

  for (int layer = 0; layer < NLAYERS; ++layer) {
    const int p = layer & 1;

    // ================= GEMM1: h1 = leaky(x[:,im] @ W1 + b1) =================
    {
      const __bf16* W1l = W1p + (size_t)layer * NMASK * HDIM;
      const int nt0 = wave * 2, nt1 = wave * 2 + 1;
      v8f acc0 = {}, acc1 = {};
      unsigned int* xmb32 = (unsigned int*)xmb;

      for (int khalf = 0; khalf < 2; ++khalf) {
        // --- cooperatively stage 32 masked+converted A-fragments ---
        for (int pairIdx = tid; pairIdx < (32 * 512) / 2; pairIdx += 256) {
          int i      = pairIdx << 1;
          int ktL    = i >> 9;
          int within = i & 511;
          int lw     = within >> 4;
          int e      = within & 15;
          int arow   = lw & 15;
          int kbw    = (lw >> 4) << 3;
          int kg     = ((khalf * 32 + ktL) << 5) + kbw + (e < 8 ? e : e + 8);
          int r0 = kg >> 5, c0 = kg & 31;
          float f0 = xs[arow * NDIM + (r0 << 6) + 2 * c0 + ((r0 + p) & 1)];
          int kg1 = kg + 1;
          int r1 = kg1 >> 5, c1 = kg1 & 31;
          float f1 = xs[arow * NDIM + (r1 << 6) + 2 * c1 + ((r1 + p) & 1)];
          xmb32[pairIdx] = pack_bf16x2(f0, f1);
        }
        __syncthreads();

        // --- 32 k-tiles of WMMA; one A load feeds both n-tiles ---
        for (int ktL = 0; ktL < 32; ++ktL) {
          int ktG = khalf * 32 + ktL;
          v16bf a = *(const v16bf*)(xmb + (ktL << 9) + (lane << 4));
          const __bf16* bp0 = W1l + ((size_t)(ktG * 16 + nt0) << 9) + (lane << 4);
          if (ktL + 1 < 32)
            __builtin_prefetch(W1l + ((size_t)((ktG + 1) * 16 + nt0) << 9) + (lane << 4), 0, 0);
          v16bf bf0 = *(const v16bf*)bp0;
          v16bf bf1 = *(const v16bf*)(W1l + ((size_t)(ktG * 16 + nt1) << 9) + (lane << 4));
          acc0 = WMMA_BF16(a, bf0, acc0);
          acc1 = WMMA_BF16(a, bf1, acc1);
        }
        __syncthreads();
      }

      float bias0 = b1[layer * HDIM + (nt0 << 4) + col];
      float bias1 = b1[layer * HDIM + (nt1 << 4) + col];
#pragma unroll
      for (int g = 0; g < 8; ++g) {
        float v0 = acc0[g] + bias0;
        v0 = v0 > 0.0f ? v0 : 0.2f * v0;
        h1[(g + rowhi) * HDIM + (nt0 << 4) + col] = (__bf16)v0;
        float v1 = acc1[g] + bias1;
        v1 = v1 > 0.0f ? v1 : 0.2f * v1;
        h1[(g + rowhi) * HDIM + (nt1 << 4) + col] = (__bf16)v1;
      }
    }
    __syncthreads();

    // ================= GEMM2: h2 = leaky(h1 @ W2 + b2) ======================
    {
      const __bf16* W2l = W2p + (size_t)layer * HDIM * HDIM;
      const int nt0 = wave * 2, nt1 = wave * 2 + 1;
      v16bf a1[8];
#pragma unroll
      for (int kt = 0; kt < 8; ++kt) a1[kt] = load_a_lds(h1, HDIM, col, kb, kt);
      v8f acc0 = {}, acc1 = {};
#pragma unroll
      for (int kt = 0; kt < 8; ++kt) {
        v16bf bf0 = *(const v16bf*)(W2l + ((size_t)(kt * 16 + nt0) << 9) + (lane << 4));
        v16bf bf1 = *(const v16bf*)(W2l + ((size_t)(kt * 16 + nt1) << 9) + (lane << 4));
        acc0 = WMMA_BF16(a1[kt], bf0, acc0);
        acc1 = WMMA_BF16(a1[kt], bf1, acc1);
      }
      float bias0 = b2[layer * HDIM + (nt0 << 4) + col];
      float bias1 = b2[layer * HDIM + (nt1 << 4) + col];
#pragma unroll
      for (int g = 0; g < 8; ++g) {
        float v0 = acc0[g] + bias0;
        v0 = v0 > 0.0f ? v0 : 0.2f * v0;
        h2[(g + rowhi) * HDIM + (nt0 << 4) + col] = (__bf16)v0;
        float v1 = acc1[g] + bias1;
        v1 = v1 > 0.0f ? v1 : 0.2f * v1;
        h2[(g + rowhi) * HDIM + (nt1 << 4) + col] = (__bf16)v1;
      }
    }
    __syncthreads();

    // ===== GEMM3: st = h2 @ W3 + b3 ; A-fragments cached in registers =====
    {
      const __bf16* W3l = W3p + (size_t)layer * HDIM * 2 * NMASK;
      const int NT3 = (2 * NMASK) / 16;  // 256
      v16bf a2[8];
#pragma unroll
      for (int kt = 0; kt < 8; ++kt) a2[kt] = load_a_lds(h2, HDIM, col, kb, kt);

      // --- s-half: s = 2*tanh(.), x[:,iu] *= exp(s), logdet += s ---
      for (int j = 0; j < 16; ++j) {
        const int nt = wave * 16 + j;   // 0..127 -> s columns
        v8f acc = {};
#pragma unroll
        for (int kt = 0; kt < 8; ++kt) {
          v16bf bfrag = *(const v16bf*)(W3l + ((size_t)(kt * NT3 + nt) << 9) + (lane << 4));
          acc = WMMA_BF16(a2[kt], bfrag, acc);
        }
        float bias = b3[layer * 2 * NMASK + (nt << 4) + col];
        int kcol = (nt << 4) + col;     // masked index 0..2047
        int r = kcol >> 5, c = kcol & 31;
        int jcol = (r << 6) + 2 * c + ((r + p + 1) & 1);   // iu position
#pragma unroll
        for (int g = 0; g < 8; ++g) {
          float s = 2.0f * tanhf(acc[g] + bias);
          ldp[g] += s;
          xs[(g + rowhi) * NDIM + jcol] *= __expf(s);
        }
      }
      __syncthreads();

      // --- t-half: x[:,iu] += t (A fragments reused from registers) ---
      for (int j = 0; j < 16; ++j) {
        const int nt = 128 + wave * 16 + j;  // 128..255 -> t columns
        v8f acc = {};
#pragma unroll
        for (int kt = 0; kt < 8; ++kt) {
          v16bf bfrag = *(const v16bf*)(W3l + ((size_t)(kt * NT3 + nt) << 9) + (lane << 4));
          acc = WMMA_BF16(a2[kt], bfrag, acc);
        }
        float bias = b3[layer * 2 * NMASK + (nt << 4) + col];
        int kcol = ((nt - 128) << 4) + col;
        int r = kcol >> 5, c = kcol & 31;
        int jcol = (r << 6) + 2 * c + ((r + p + 1) & 1);
#pragma unroll
        for (int g = 0; g < 8; ++g) {
          xs[(g + rowhi) * NDIM + jcol] += acc[g] + bias;
        }
      }
      __syncthreads();
    }
  }

  // ---- log-det reduction (ds_add_f32) and write-back ----
#pragma unroll
  for (int g = 0; g < 8; ++g) atomicAdd(&ldacc[g + rowhi], ldp[g]);
  __syncthreads();

  {
    const float4* src = (const float4*)xs;
    float4*       dst = (float4*)(x_out + (size_t)b0 * NDIM);
    for (int it = tid; it < TILE_B * NDIM / 4; it += 256) dst[it] = src[it];
  }
  if (tid < TILE_B) ld_out[b0 + tid] = ldacc[tid];
}

// ---------------------------------------------------------------------------
extern "C" void kernel_launch(void* const* d_in, const int* in_sizes, int n_in,
                              void* d_out, int out_size, void* d_ws, size_t ws_size,
                              hipStream_t stream) {
  const float* z  = (const float*)d_in[0];
  const float* W1 = (const float*)d_in[1];
  const float* b1 = (const float*)d_in[2];
  const float* W2 = (const float*)d_in[3];
  const float* b2 = (const float*)d_in[4];
  const float* W3 = (const float*)d_in[5];
  const float* b3 = (const float*)d_in[6];

  unsigned short* W1p = (unsigned short*)d_ws;
  unsigned short* W2p = W1p + (size_t)NLAYERS * NMASK * HDIM;
  unsigned short* W3p = W2p + (size_t)NLAYERS * HDIM * HDIM;

  {
    long long t1 = (long long)NLAYERS * NMASK * HDIM;
    pack_weights<<<(unsigned)((t1 + 255) / 256), 256, 0, stream>>>(W1, W1p, NLAYERS, NMASK, HDIM);
    long long t2 = (long long)NLAYERS * HDIM * HDIM;
    pack_weights<<<(unsigned)((t2 + 255) / 256), 256, 0, stream>>>(W2, W2p, NLAYERS, HDIM, HDIM);
    long long t3 = (long long)NLAYERS * HDIM * 2 * NMASK;
    pack_weights<<<(unsigned)((t3 + 255) / 256), 256, 0, stream>>>(W3, W3p, NLAYERS, HDIM, 2 * NMASK);
  }

  size_t lds_bytes = (size_t)TILE_B * NDIM * 4         // x tile fp32
                   + (size_t)32 * 512 * 2              // A-fragment staging (bf16)
                   + (size_t)2 * TILE_B * HDIM * 2     // h1, h2 bf16
                   + (size_t)TILE_B * 4;               // logdet accum
  float* x_out  = (float*)d_out;
  float* ld_out = x_out + (size_t)BATCH * NDIM;

  realnvp_flow<<<BATCH / TILE_B, 256, lds_bytes, stream>>>(
      z, b1, b2, b3, W1p, W2p, W3p, x_out, ld_out);
}
